// Swin_5669356833041
// MI455X (gfx1250) — matrix operand order repaired and checked by reading, not compiled, not run on hardware
//
#include <hip/hip_runtime.h>
#include <hip/hip_bf16.h>

typedef _Float16 h16;
typedef __attribute__((ext_vector_type(16))) _Float16 v16h;
typedef __attribute__((ext_vector_type(8)))  float    v8f;
typedef __attribute__((ext_vector_type(4)))  unsigned int u32x4;
typedef __attribute__((ext_vector_type(8)))  int      i32x8;
typedef __attribute__((ext_vector_type(4)))  int      i32x4;

#define HEADS 8
#define DH    32
#define PP    49
#define PPAD  64
#define CDIM  256
#define NQKV  768
#define HW    56
#define DSH   3

#if defined(__has_builtin)
#if __has_builtin(__builtin_amdgcn_tensor_load_to_lds) && defined(__AMDGCN__)
#define HAVE_TDM 1
#endif
#if __has_builtin(__builtin_amdgcn_sched_barrier)
#define SCHED_FENCE() __builtin_amdgcn_sched_barrier(0)
#endif
#endif
#ifndef SCHED_FENCE
#define SCHED_FENCE() ((void)0)
#endif

// ---------- WMMA fragment loaders (CDNA5 16x16x32 f16 layouts) ----------

// A fragment: rows m0..m0+15, K k0..k0+31 of a row-major [*, ld] f16 matrix.
// Lane layout: row = lane&15, half h = lane>>4; elements 0..15 -> K = h*16+i
// (contiguous 32B run).
__device__ __forceinline__ v16h load_a(const h16* p, int m0, int k0, int ld) {
  int l = threadIdx.x & 31;
  int r = l & 15, h = l >> 4;
  const h16* q = p + (size_t)(m0 + r) * ld + k0;
  v16h out;
#pragma unroll
  for (int i = 0; i < 16; ++i) out[i] = q[h * 16 + i];
  return out;
}

// B fragment, loaded from B^T stored row-major [N][K]: B columns n0..n0+15,
// K k0..k0+31. Lane layout: col = lane&15, K = h*16 + i (contiguous 32B run).
__device__ __forceinline__ v16h load_bT(const h16* pT, int n0, int k0, int ld) {
  int l = threadIdx.x & 31;
  int n = l & 15, h = l >> 4;
  const h16* q = pT + (size_t)(n0 + n) * ld + k0 + h * 16;
  v16h out;
#pragma unroll
  for (int i = 0; i < 16; ++i) out[i] = q[i];
  return out;
}

// B fragment from B stored row-major [K][N] (strided per-element reads).
__device__ __forceinline__ v16h load_b_rm(const h16* p, int k0, int n0, int ld) {
  int l = threadIdx.x & 31;
  int n = l & 15, h = l >> 4;
  v16h out;
#pragma unroll
  for (int i = 0; i < 16; ++i) out[i] = p[(size_t)(k0 + h * 16 + i) * ld + n0 + n];
  return out;
}

__device__ __forceinline__ v8f wmma_f16(v16h a, v16h b, v8f c) {
  return __builtin_amdgcn_wmma_f32_16x16x32_f16(false, a, false, b, (short)0, c,
                                                false, false);
}

// ---------- Tensor Data Mover: 2D tile (16-bit elements) global -> LDS ----------
// Builds D# groups per CDNA5 ISA. tensor dims == tile dims
// (tile start address is the descriptor's global_addr, so no OOB clipping).
// stride_d0 in elements. Caller must be a single wave; TDM ignores EXEC.
#ifdef HAVE_TDM
__device__ __forceinline__ void tdm_load_2d_b16(unsigned lds_byte,
                                                const void* gaddr,
                                                unsigned tile_d0, unsigned tile_d1,
                                                unsigned stride_d0) {
  unsigned long long ga = (unsigned long long)gaddr;
  u32x4 g0;
  g0[0] = 1u;                                           // count=1, user D#
  g0[1] = lds_byte;                                     // lds_addr [63:32]
  g0[2] = (unsigned)(ga & 0xffffffffu);                 // global_addr lo
  g0[3] = (unsigned)((ga >> 32) & 0x01ffffffu) | (2u << 30);  // hi + type=2
  i32x8 g1;
  g1[0] = (int)(1u << 16);                              // data_size=2B, mask=0
  g1[1] = (int)((tile_d0 & 0xffffu) << 16);             // tensor_dim0 lo16
  g1[2] = (int)(((tile_d0 >> 16) & 0xffffu) |           // tensor_dim0 hi16
                ((tile_d1 & 0xffffu) << 16));           // tensor_dim1 lo16
  g1[3] = (int)(((tile_d1 >> 16) & 0xffffu) |           // tensor_dim1 hi16
                ((tile_d0 & 0xffffu) << 16));           // tile_dim0
  g1[4] = (int)(tile_d1 & 0xffffu);                     // tile_dim1 (tile_dim2=0)
  g1[5] = (int)stride_d0;                               // tensor_dim0_stride lo32
  g1[6] = 0;
  g1[7] = 0;
  i32x4 gz = {};
#if __clang_major__ >= 23
  i32x8 gz8 = {};
  __builtin_amdgcn_tensor_load_to_lds(g0, g1, gz, gz, gz8, 0);
#else
  __builtin_amdgcn_tensor_load_to_lds(g0, g1, gz, gz, 0);
#endif
}
#endif

// ---------- prolog: transpose weights to f16 [N][K] ----------
__global__ void transpose_to_f16(const float* __restrict__ src,
                                 h16* __restrict__ dst, int K, int N) {
  int idx = blockIdx.x * blockDim.x + threadIdx.x;
  if (idx < K * N) {
    int n = idx / K, k = idx % K;
    dst[idx] = (h16)src[k * N + n];   // dst[n][k] = src[k][n]
  }
}

// ---------- stage A: window gather + QKV GEMM (49x256 @ 256x768) ----------
__global__ __launch_bounds__(256) void qkv_kernel(const float* __restrict__ x,
                                                  const h16* __restrict__ wqkvT,
                                                  h16* __restrict__ qkv16) {
  __shared__ h16 sA[PPAD * CDIM];     // 64 x 256 f16 window (rows >=49 zero)
  int bn = blockIdx.x;
  int batch = bn >> 6, n = bn & 63;
  int nr = n >> 3, nc = n & 7;
  int tid = threadIdx.x;

  for (int i = 0; i < PP; ++i) {      // channel = tid; roll(-3,-3) gather
    int hs = (nr * 7 + i / 7 + DSH) % HW;
    int ws = (nc * 7 + i % 7 + DSH) % HW;
    float v = x[((size_t)(batch * CDIM + tid) * HW + hs) * HW + ws];
    sA[i * CDIM + tid] = (h16)v;
  }
#pragma unroll
  for (int k = 0; k < PPAD - PP; ++k) sA[(PP + k) * CDIM + tid] = (h16)0.f;
  __syncthreads();

  int wave = tid >> 5, l = tid & 31, lc = l & 15, lh = l >> 4;
  int mt = wave & 3;                  // wave-constant M tile
  h16* outBase = qkv16 + (size_t)bn * PP * NQKV;

  v16h afr[8];                        // A fragments resident across all N tiles
#pragma unroll
  for (int ks = 0; ks < 8; ++ks) afr[ks] = load_a(sA, mt * 16, ks * 32, CDIM);

  for (int j = 0; j < 24; ++j) {      // waves 0-3: even nt, waves 4-7: odd nt
    int nt = (wave >> 2) + 2 * j;
    v16h bfr[8];                      // cluster all 16 b128 loads up front
#pragma unroll
    for (int ks = 0; ks < 8; ++ks) bfr[ks] = load_bT(wqkvT, nt * 16, ks * 32, CDIM);
    SCHED_FENCE();                    // keep loads ahead of the WMMA chain
    v8f acc = {};
#pragma unroll
    for (int ks = 0; ks < 8; ++ks) acc = wmma_f16(afr[ks], bfr[ks], acc);
    int col = nt * 16 + lc;
#pragma unroll
    for (int r = 0; r < 8; ++r) {
      int row = mt * 16 + r + 8 * lh;
      if (row < PP) outBase[(size_t)row * NQKV + col] = (h16)acc[r];
    }
  }
}

// ---------- stage B: per (window, head) attention ----------
__global__ __launch_bounds__(128) void attn_kernel(const h16* __restrict__ qkv16,
                                                   const float* __restrict__ pe,
                                                   float* __restrict__ attn_out,
                                                   h16* __restrict__ o16) {
  __shared__ h16 sQ[PPAD * DH], sK[PPAD * DH], sV[PPAD * DH];
  __shared__ float sS[PPAD * PPAD];
  __shared__ h16 sP[PPAD * PPAD];
  __shared__ float sPE[169];
  int bn = blockIdx.x, head = blockIdx.y;
  int n = bn & 63;
  bool ul = (n >= 56);                 // last row of windows: mask on h coord
  bool lr = (n == 55) || (n == 63);    // reference's [-n1-1::n1] slice
  int tid = threadIdx.x;

  const h16* base = qkv16 + (size_t)bn * PP * NQKV + head * DH;
#ifdef HAVE_TDM
  // zero the pad rows (49..63) with all threads; TDM fills rows 0..48.
  for (int e = PP * DH + tid; e < PPAD * DH; e += 128) {
    sQ[e] = (h16)0.f; sK[e] = (h16)0.f; sV[e] = (h16)0.f;
  }
  if (tid < 32) {                      // wave 0 issues the three 2D DMAs
    tdm_load_2d_b16((unsigned)(size_t)&sQ[0], base,        DH, PP, NQKV);
    tdm_load_2d_b16((unsigned)(size_t)&sK[0], base + 256,  DH, PP, NQKV);
    tdm_load_2d_b16((unsigned)(size_t)&sV[0], base + 512,  DH, PP, NQKV);
    __builtin_amdgcn_s_wait_tensorcnt(0);
  }
#else
  for (int e = tid; e < PPAD * DH; e += 128) {
    int i = e >> 5, c = e & 31;
    h16 q = (h16)0.f, k = (h16)0.f, v = (h16)0.f;
    if (i < PP) {
      q = base[(size_t)i * NQKV + c];
      k = base[(size_t)i * NQKV + 256 + c];
      v = base[(size_t)i * NQKV + 512 + c];
    }
    sQ[e] = q; sK[e] = k; sV[e] = v;
  }
#endif
  for (int e = tid; e < 169; e += 128) sPE[e] = pe[e];
  __syncthreads();

  int wave = tid >> 5, l = tid & 31, lc = l & 15, lh = l >> 4;
  const float scale = 0.17677669529663687f;   // 32^-0.5
  // S = Q @ K^T : wave owns M-tile `wave`; 4 N-tiles; K=32 -> single WMMA.
  v16h aq = load_a(sQ, wave * 16, 0, DH);
#pragma unroll
  for (int nt = 0; nt < 4; ++nt) {
    v16h b = load_bT(sK, nt * 16, 0, DH);    // K^T columns = K rows
    v8f acc = {};
    acc = wmma_f16(aq, b, acc);
    int j = nt * 16 + lc;
    int hj = j / 7, wj = j % 7;
#pragma unroll
    for (int r = 0; r < 8; ++r) {
      int i = wave * 16 + r + 8 * lh;
      float s = -1e30f;
      if (i < PP && j < PP) {
        int hi = i / 7, wi = i % 7;
        s = acc[r] * scale + sPE[(hj - hi + 6) * 13 + (wj - wi + 6)];
        if (ul && ((hi >= 4) != (hj >= 4))) s = -1e30f;
        if (lr && ((wi >= 4) != (wj >= 4))) s = -1e30f;
      }
      sS[i * PPAD + j] = s;
    }
  }
  __syncthreads();

  // row softmax (f32), emit attn (f32 output) and P (f16 operand for P@V)
  if (tid < PPAD) {
    int i = tid;
    float m = -1e30f;
    for (int j = 0; j < PP; ++j) m = fmaxf(m, sS[i * PPAD + j]);
    float sum = 0.f;
    for (int j = 0; j < PP; ++j) sum += __expf(sS[i * PPAD + j] - m);
    float inv = 1.f / sum;
    float* ao = attn_out + ((size_t)(bn * HEADS + head) * PP + i) * PP;
    for (int j = 0; j < PPAD; ++j) {
      float p = 0.f;
      if (i < PP && j < PP) {
        p = __expf(sS[i * PPAD + j] - m) * inv;
        ao[j] = p;
      }
      sP[i * PPAD + j] = (h16)p;
    }
  }
  __syncthreads();

  // O = P(64x64) @ V(64x32): wave -> M-tile, 2 N-tiles, 2 K-steps.
  h16* ob = o16 + (size_t)bn * PP * CDIM + head * DH;
#pragma unroll
  for (int nt = 0; nt < 2; ++nt) {
    v8f acc = {};
#pragma unroll
    for (int ks = 0; ks < 2; ++ks) {
      v16h a = load_a(sP, wave * 16, ks * 32, PPAD);
      v16h b = load_b_rm(sV, ks * 32, nt * 16, DH);
      acc = wmma_f16(a, b, acc);
    }
    int col = nt * 16 + lc;
#pragma unroll
    for (int r = 0; r < 8; ++r) {
      int i = wave * 16 + r + 8 * lh;
      if (i < PP) ob[(size_t)i * CDIM + col] = (h16)acc[r];
    }
  }
}

// ---------- stage C: out projection + inverse window transform + roll ----------
__global__ __launch_bounds__(256) void proj_kernel(const h16* __restrict__ o16,
                                                   const h16* __restrict__ woutT,
                                                   const float* __restrict__ b_out,
                                                   float* __restrict__ out) {
  __shared__ h16 sO[PPAD * CDIM];
  int bn = blockIdx.x;
  int batch = bn >> 6, n = bn & 63;
  int nr = n >> 3, nc = n & 7;
  int tid = threadIdx.x;

  const h16* ib = o16 + (size_t)bn * PP * CDIM;
#ifdef HAVE_TDM
  for (int e = PP * CDIM + tid; e < PPAD * CDIM; e += 256) sO[e] = (h16)0.f;
  if (tid < 32) {                      // one 2D DMA: 256x49 halves, stride 256
    tdm_load_2d_b16((unsigned)(size_t)&sO[0], ib, CDIM, PP, CDIM);
    __builtin_amdgcn_s_wait_tensorcnt(0);
  }
#else
  for (int e = tid; e < PPAD * CDIM; e += 256) {
    int i = e >> 8;
    sO[e] = (i < PP) ? ib[e] : (h16)0.f;
  }
#endif
  __syncthreads();

  int wave = tid >> 5, l = tid & 31, lc = l & 15, lh = l >> 4;
  int mt = wave & 3;
  v16h afr[8];
#pragma unroll
  for (int ks = 0; ks < 8; ++ks) afr[ks] = load_a(sO, mt * 16, ks * 32, CDIM);

  for (int j = 0; j < 8; ++j) {        // 16 N tiles over 2 wave groups
    int nt = (wave >> 2) + 2 * j;
    v16h bfr[8];
#pragma unroll
    for (int ks = 0; ks < 8; ++ks) bfr[ks] = load_bT(woutT, nt * 16, ks * 32, CDIM);
    SCHED_FENCE();                     // keep loads ahead of the WMMA chain
    v8f acc = {};
#pragma unroll
    for (int ks = 0; ks < 8; ++ks) acc = wmma_f16(afr[ks], bfr[ks], acc);
    int c = nt * 16 + lc;
    float bo = b_out[c];
#pragma unroll
    for (int r = 0; r < 8; ++r) {
      int i = mt * 16 + r + 8 * lh;
      if (i < PP) {
        int ho = (nr * 7 + i / 7 + DSH) % HW;   // roll(+3,+3)
        int wo = (nc * 7 + i % 7 + DSH) % HW;
        out[((size_t)(batch * CDIM + c) * HW + ho) * HW + wo] = acc[r] + bo;
      }
    }
  }
}

extern "C" void kernel_launch(void* const* d_in, const int* in_sizes, int n_in,
                              void* d_out, int out_size, void* d_ws, size_t ws_size,
                              hipStream_t stream) {
  const float* x      = (const float*)d_in[0];   // 32x256x56x56
  const float* pe     = (const float*)d_in[1];   // 13x13
  const float* w_qkv  = (const float*)d_in[2];   // 256x768
  const float* w_out  = (const float*)d_in[3];   // 256x256
  const float* b_out  = (const float*)d_in[4];   // 256
  float* out  = (float*)d_out;                           // 25,690,112 floats
  float* attn = out + (size_t)32 * 256 * 56 * 56;        // 2048x8x49x49

  h16* wqkvT = (h16*)d_ws;                               // [768][256]
  h16* woutT = wqkvT + (size_t)768 * 256;                // [256][256]
  h16* qkv16 = woutT + (size_t)256 * 256;                // [2048][49][768]
  h16* o16   = qkv16 + (size_t)2048 * PP * NQKV;         // [2048][49][256]

  transpose_to_f16<<<(768 * 256 + 255) / 256, 256, 0, stream>>>(w_qkv, wqkvT, 256, 768);
  transpose_to_f16<<<(256 * 256 + 255) / 256, 256, 0, stream>>>(w_out, woutT, 256, 256);
  qkv_kernel<<<2048, 256, 0, stream>>>(x, wqkvT, qkv16);
  attn_kernel<<<dim3(2048, 8), 128, 0, stream>>>(qkv16, pe, attn, o16);
  proj_kernel<<<2048, 256, 0, stream>>>(o16, woutT, b_out, out);
}
